// SCARERNNAgent_64201171141435
// MI455X (gfx1250) — matrix operand order repaired and checked by reading, not compiled, not run on hardware
//
#include <hip/hip_runtime.h>
#include <hip/hip_bf16.h>

// ---------------------------------------------------------------------------
// Problem constants (from reference): B, D, H, K, E, HS, A
// ---------------------------------------------------------------------------
#define NB    8192
#define ND    512
#define NH    1024
#define NK    8
#define NE    1024
#define NHS   256
#define NA    32

typedef __attribute__((ext_vector_type(16))) __bf16 v16bf;
typedef __attribute__((ext_vector_type(8)))  __bf16 v8bf;
typedef __attribute__((ext_vector_type(8)))  float  v8f;

// fp32 -> bf16 round-to-nearest-even (pure integer ops)
__device__ __forceinline__ __bf16 f2bf(float f) {
  union { float f; unsigned u; } a; a.f = f;
  unsigned r = a.u + 0x7FFFu + ((a.u >> 16) & 1u);
  union { unsigned short s; __bf16 b; } o;
  o.s = (unsigned short)(r >> 16);
  return o.b;
}
__device__ __forceinline__ float bf2f(__bf16 b) {
  union { unsigned short s; __bf16 b; } i; i.b = b;
  union { unsigned u; float f; } o; o.u = ((unsigned)i.s) << 16;
  return o.f;
}

// LDS tile stride: 40 bf16 (= 80 B) keeps 16B alignment, spreads banks.
#define TS 40

// A-fragment (16x32 bf16, M x K): lane 0-15 row M=lane holds K=[0..7],[16..23];
// lanes 16-31 hold K=[8..15],[24..31].  (ISA 7.12.2)
__device__ __forceinline__ v16bf ld_frag_a(const __bf16* tile, int row0, int lane) {
  const __bf16* p = tile + (size_t)(row0 + (lane & 15)) * TS;
  int h = (lane >> 4) * 8;
  union { v16bf v; v8bf q[2]; } u;
  u.q[0] = *(const v8bf*)(p + h);
  u.q[1] = *(const v8bf*)(p + 16 + h);
  return u.v;
}
// B-fragment (32x16 bf16, K x N), tile stored [N][K] row-major:
// lane = column N; lanes 0-15 hold K=[0..7],[16..23]; lanes 16-31 K=[8..15],[24..31].
__device__ __forceinline__ v16bf ld_frag_b(const __bf16* tile, int row0, int lane) {
  const __bf16* p = tile + (size_t)(row0 + (lane & 15)) * TS + (lane >> 4) * 8;
  union { v16bf v; v8bf q[2]; } u;
  u.q[0] = *(const v8bf*)(p);
  u.q[1] = *(const v8bf*)(p + 16);
  return u.v;
}

__device__ __forceinline__ v8f wmma_bf16(v16bf a, v16bf b, v8f c) {
  // (neg_a, A, neg_b, B, c_mod, C, reuse_a, reuse_b)
  return __builtin_amdgcn_wmma_f32_16x16x32_bf16(false, a, false, b, (short)0, c,
                                                 false, false);
}

__device__ __forceinline__ float sigmoidf_(float x) {
  return 1.0f / (1.0f + __expf(-x));
}

// ---------------------------------------------------------------------------
// Generic WMMA GEMM: C = act(A[M,Kd] * W[N,Kd]^T + bias[N])
// BM=128, BN=64, BK=32; 256 threads = 8 waves (4x2), wave tile 32x32.
// grid = (ceil(N/64), M/128, Z); per-z strides for batched weights/outputs.
// ---------------------------------------------------------------------------
template <bool RELU, bool OUT_BF16>
__global__ __launch_bounds__(256) void gemm_wmma_kernel(
    const float* __restrict__ A, int lda,
    const float* __restrict__ W, const float* __restrict__ bias,
    void* __restrict__ C, int ldc, int M, int N, int Kd,
    long wStride, long bStride, long cBatchOff) {
  __shared__ __bf16 As[128 * TS];
  __shared__ __bf16 Bs[64 * TS];

  const int tid  = threadIdx.x;
  const int lane = tid & 31;
  const int wave = tid >> 5;
  const int wr   = wave >> 1;   // 0..3 -> 32 rows each
  const int wc   = wave & 1;    // 0..1 -> 32 cols each
  const int m0   = blockIdx.y * 128;
  const int n0   = blockIdx.x * 64;
  const int z    = blockIdx.z;
  const float* Wz = W + (size_t)z * wStride;
  const float* bz = bias + (size_t)z * bStride;

  v8f zero = {0.f, 0.f, 0.f, 0.f, 0.f, 0.f, 0.f, 0.f};
  v8f acc[2][2] = {{zero, zero}, {zero, zero}};

  for (int kt = 0; kt < Kd; kt += 32) {
    // A tile: 128x32 fp32 -> bf16 LDS (4 float4 per thread)
#pragma unroll
    for (int i = 0; i < 4; ++i) {
      int v = tid + i * 256;
      int row = v >> 3, kc = (v & 7) * 4;
      float4 f = *(const float4*)(A + (size_t)(m0 + row) * lda + kt + kc);
      __bf16* d = &As[row * TS + kc];
      d[0] = f2bf(f.x); d[1] = f2bf(f.y); d[2] = f2bf(f.z); d[3] = f2bf(f.w);
    }
    // W tile: 64x32 (row-major [N][Kd]) -> LDS, guard partial N
#pragma unroll
    for (int i = 0; i < 2; ++i) {
      int v = tid + i * 256;
      int row = v >> 3, kc = (v & 7) * 4;
      float4 f = {0.f, 0.f, 0.f, 0.f};
      if (n0 + row < N)
        f = *(const float4*)(Wz + (size_t)(n0 + row) * Kd + kt + kc);
      __bf16* d = &Bs[row * TS + kc];
      d[0] = f2bf(f.x); d[1] = f2bf(f.y); d[2] = f2bf(f.z); d[3] = f2bf(f.w);
    }
    __syncthreads();

    v16bf a0 = ld_frag_a(As, wr * 32, lane);
    v16bf a1 = ld_frag_a(As, wr * 32 + 16, lane);
    v16bf b0 = ld_frag_b(Bs, wc * 32, lane);
    v16bf b1 = ld_frag_b(Bs, wc * 32 + 16, lane);
    acc[0][0] = wmma_bf16(a0, b0, acc[0][0]);
    acc[0][1] = wmma_bf16(a0, b1, acc[0][1]);
    acc[1][0] = wmma_bf16(a1, b0, acc[1][0]);
    acc[1][1] = wmma_bf16(a1, b1, acc[1][1]);
    __syncthreads();
  }

  const size_t cbase = (size_t)z * cBatchOff;
#pragma unroll
  for (int r = 0; r < 2; ++r)
#pragma unroll
    for (int c = 0; c < 2; ++c)
#pragma unroll
      for (int j = 0; j < 8; ++j) {
        int m = ((lane >> 4) << 3) + j;
        int n = lane & 15;
        int gm = m0 + wr * 32 + r * 16 + m;
        int gn = n0 + wc * 32 + c * 16 + n;
        if (gn < N && gm < M) {
          float val = acc[r][c][j] + bz[gn];
          if (RELU) val = fmaxf(val, 0.f);
          if (OUT_BF16)
            ((__bf16*)C)[cbase + (size_t)gm * ldc + gn] = f2bf(val);
          else
            ((float*)C)[cbase + (size_t)gm * ldc + gn] = val;
        }
      }
}

// ---------------------------------------------------------------------------
// Selector logits + softmax: alpha[b,k] = softmax_k(sh[b,:] . Ws2[k,:] + bs2)
// one wave per row, lane-parallel over HS=256.
// ---------------------------------------------------------------------------
__global__ __launch_bounds__(256) void selector_kernel(
    const void* __restrict__ sh_, const float* __restrict__ Ws2,
    const float* __restrict__ bs2, float* __restrict__ alpha) {
  const __bf16* sh = (const __bf16*)sh_;
  int lane = threadIdx.x & 31;
  int wave = threadIdx.x >> 5;
  int b = blockIdx.x * 8 + wave;

  float logit[NK];
#pragma unroll
  for (int kk = 0; kk < NK; ++kk) {
    float p = 0.f;
    for (int i = lane; i < NHS; i += 32)
      p += bf2f(sh[(size_t)b * NHS + i]) * Ws2[kk * NHS + i];
#pragma unroll
    for (int off = 16; off; off >>= 1) p += __shfl_xor(p, off);
    logit[kk] = p + bs2[kk];
  }
  float mx = logit[0];
#pragma unroll
  for (int kk = 1; kk < NK; ++kk) mx = fmaxf(mx, logit[kk]);
  float s = 0.f;
#pragma unroll
  for (int kk = 0; kk < NK; ++kk) s += __expf(logit[kk] - mx);
  if (lane < NK) {
    float lv = logit[0];
#pragma unroll
    for (int kk = 1; kk < NK; ++kk)
      if (lane == kk) lv = logit[kk];
    alpha[(size_t)b * NK + lane] = __expf(lv - mx) / s;
  }
}

// ---------------------------------------------------------------------------
// Fused GRU: for branch k, tile (64 batch x 64 h-cols):
//   six WMMA accumulators (gi_r/z/n from x_bf16, gh_r/z/n from hidden fp32),
//   then gate math -> h_out.  grid = (H/64, B/64, K); wave tile 16x32.
// ---------------------------------------------------------------------------
__global__ __launch_bounds__(256) void gru_fused_kernel(
    const void* __restrict__ x_, const float* __restrict__ hidden,
    const float* __restrict__ Wih, const float* __restrict__ Whh,
    const float* __restrict__ bih, const float* __restrict__ bhh,
    float* __restrict__ h_out) {
  const __bf16* xb = (const __bf16*)x_;
  __shared__ __bf16 Xs[64 * TS];
  __shared__ __bf16 Hs[64 * TS];
  __shared__ __bf16 Wt[6 * 64 * TS];

  const int tid  = threadIdx.x;
  const int lane = tid & 31;
  const int wave = tid >> 5;
  const int wr   = wave >> 1;   // 4 row groups of 16
  const int wc   = wave & 1;    // 2 col groups of 32
  const int b0   = blockIdx.y * 64;
  const int n0   = blockIdx.x * 64;
  const int k    = blockIdx.z;

  const float* Wih_k = Wih + (size_t)k * 3 * NH * NH;
  const float* Whh_k = Whh + (size_t)k * 3 * NH * NH;

  v8f zero = {0.f, 0.f, 0.f, 0.f, 0.f, 0.f, 0.f, 0.f};
  v8f ai[3][2], ah[3][2];
#pragma unroll
  for (int g = 0; g < 3; ++g)
#pragma unroll
    for (int c = 0; c < 2; ++c) { ai[g][c] = zero; ah[g][c] = zero; }

  for (int kt = 0; kt < NH; kt += 32) {
    // X tile: bf16 copy (2 x 8B per thread)
#pragma unroll
    for (int i = 0; i < 2; ++i) {
      int v = tid + i * 256;
      int row = v >> 3, kc = (v & 7) * 4;
      uint2 u = *(const uint2*)(xb + (size_t)(b0 + row) * (NK * NH) +
                                (size_t)k * NH + kt + kc);
      *(uint2*)(&Xs[row * TS + kc]) = u;
    }
    // H tile: fp32 -> bf16
#pragma unroll
    for (int i = 0; i < 2; ++i) {
      int v = tid + i * 256;
      int row = v >> 3, kc = (v & 7) * 4;
      float4 f = *(const float4*)(hidden + (size_t)(b0 + row) * (NK * NH) +
                                  (size_t)k * NH + kt + kc);
      __bf16* d = &Hs[row * TS + kc];
      d[0] = f2bf(f.x); d[1] = f2bf(f.y); d[2] = f2bf(f.z); d[3] = f2bf(f.w);
    }
    // Six 64x32 weight tiles (r,z,n for Wih then Whh)
#pragma unroll
    for (int w6 = 0; w6 < 6; ++w6) {
      const float* Ws = (w6 < 3) ? Wih_k : Whh_k;
      int g = (w6 < 3) ? w6 : (w6 - 3);
#pragma unroll
      for (int i = 0; i < 2; ++i) {
        int v = tid + i * 256;
        int row = v >> 3, kc = (v & 7) * 4;
        float4 f = *(const float4*)(Ws + (size_t)(g * NH + n0 + row) * NH + kt + kc);
        __bf16* d = &Wt[w6 * 64 * TS + row * TS + kc];
        d[0] = f2bf(f.x); d[1] = f2bf(f.y); d[2] = f2bf(f.z); d[3] = f2bf(f.w);
      }
    }
    __syncthreads();

    v16bf ax = ld_frag_a(Xs, wr * 16, lane);
    v16bf av = ld_frag_a(Hs, wr * 16, lane);
#pragma unroll
    for (int g = 0; g < 3; ++g)
#pragma unroll
      for (int c = 0; c < 2; ++c) {
        v16bf bi = ld_frag_b(&Wt[g * 64 * TS], wc * 32 + c * 16, lane);
        ai[g][c] = wmma_bf16(ax, bi, ai[g][c]);
        v16bf bh = ld_frag_b(&Wt[(g + 3) * 64 * TS], wc * 32 + c * 16, lane);
        ah[g][c] = wmma_bf16(av, bh, ah[g][c]);
      }
    __syncthreads();
  }

  // GRU gate math + store h
#pragma unroll
  for (int c = 0; c < 2; ++c)
#pragma unroll
    for (int j = 0; j < 8; ++j) {
      int m = ((lane >> 4) << 3) + j;
      int n = lane & 15;
      int gb = b0 + wr * 16 + m;
      int gn = n0 + wc * 32 + c * 16 + n;
      float gir = ai[0][c][j] + bih[k * 3 * NH + gn];
      float giz = ai[1][c][j] + bih[k * 3 * NH + NH + gn];
      float gin = ai[2][c][j] + bih[k * 3 * NH + 2 * NH + gn];
      float ghr = ah[0][c][j] + bhh[k * 3 * NH + gn];
      float ghz = ah[1][c][j] + bhh[k * 3 * NH + NH + gn];
      float ghn = ah[2][c][j] + bhh[k * 3 * NH + 2 * NH + gn];
      float r  = sigmoidf_(gir + ghr);
      float zg = sigmoidf_(giz + ghz);
      float nn = tanhf(gin + r * ghn);
      float hp = hidden[(size_t)gb * (NK * NH) + (size_t)k * NH + gn];
      h_out[(size_t)gb * (NK * NH) + (size_t)k * NH + gn] =
          (1.f - zg) * nn + zg * hp;
    }
}

// ---------------------------------------------------------------------------
// Fused combine: combined[b,e] = sum_k alpha[b,k]*(h[b,k,:] . Wenc[k,e,:] + benc[k,e])
// alpha folded into A-tile during fp32->bf16 conversion; one accumulator set.
// grid = (E/64, B/128); wave tile 32x32.
// ---------------------------------------------------------------------------
__global__ __launch_bounds__(256) void combine_kernel(
    const float* __restrict__ h, const float* __restrict__ alpha,
    const float* __restrict__ Wenc, const float* __restrict__ benc,
    float* __restrict__ combined) {
  __shared__ __bf16 As[128 * TS];
  __shared__ __bf16 Bs[64 * TS];

  const int tid  = threadIdx.x;
  const int lane = tid & 31;
  const int wave = tid >> 5;
  const int wr   = wave >> 1;
  const int wc   = wave & 1;
  const int m0   = blockIdx.y * 128;
  const int n0   = blockIdx.x * 64;

  v8f zero = {0.f, 0.f, 0.f, 0.f, 0.f, 0.f, 0.f, 0.f};
  v8f acc[2][2] = {{zero, zero}, {zero, zero}};

  for (int kb = 0; kb < NK; ++kb) {
    for (int kt = 0; kt < NH; kt += 32) {
      // A tile: alpha-scaled h rows
#pragma unroll
      for (int i = 0; i < 4; ++i) {
        int v = tid + i * 256;
        int row = v >> 3, kc = (v & 7) * 4;
        float al = alpha[(size_t)(m0 + row) * NK + kb];
        float4 f = *(const float4*)(h + (size_t)(m0 + row) * (NK * NH) +
                                    (size_t)kb * NH + kt + kc);
        __bf16* d = &As[row * TS + kc];
        d[0] = f2bf(f.x * al); d[1] = f2bf(f.y * al);
        d[2] = f2bf(f.z * al); d[3] = f2bf(f.w * al);
      }
      // B tile: Wenc[kb] rows
#pragma unroll
      for (int i = 0; i < 2; ++i) {
        int v = tid + i * 256;
        int row = v >> 3, kc = (v & 7) * 4;
        float4 f = *(const float4*)(Wenc + ((size_t)kb * NE + n0 + row) * NH + kt + kc);
        __bf16* d = &Bs[row * TS + kc];
        d[0] = f2bf(f.x); d[1] = f2bf(f.y); d[2] = f2bf(f.z); d[3] = f2bf(f.w);
      }
      __syncthreads();

      v16bf a0 = ld_frag_a(As, wr * 32, lane);
      v16bf a1 = ld_frag_a(As, wr * 32 + 16, lane);
      v16bf b0 = ld_frag_b(Bs, wc * 32, lane);
      v16bf b1 = ld_frag_b(Bs, wc * 32 + 16, lane);
      acc[0][0] = wmma_bf16(a0, b0, acc[0][0]);
      acc[0][1] = wmma_bf16(a0, b1, acc[0][1]);
      acc[1][0] = wmma_bf16(a1, b0, acc[1][0]);
      acc[1][1] = wmma_bf16(a1, b1, acc[1][1]);
      __syncthreads();
    }
  }

#pragma unroll
  for (int r = 0; r < 2; ++r)
#pragma unroll
    for (int c = 0; c < 2; ++c)
#pragma unroll
      for (int j = 0; j < 8; ++j) {
        int m = ((lane >> 4) << 3) + j;
        int n = lane & 15;
        int gm = m0 + wr * 32 + r * 16 + m;
        int gn = n0 + wc * 32 + c * 16 + n;
        float val = acc[r][c][j];
#pragma unroll
        for (int kk = 0; kk < NK; ++kk)
          val += alpha[(size_t)gm * NK + kk] * benc[kk * NE + gn];
        combined[(size_t)gm * NE + gn] = val;
      }
}

// ---------------------------------------------------------------------------
// Host-side launcher
// ---------------------------------------------------------------------------
extern "C" void kernel_launch(void* const* d_in, const int* in_sizes, int n_in,
                              void* d_out, int out_size, void* d_ws, size_t ws_size,
                              hipStream_t stream) {
  (void)in_sizes; (void)n_in; (void)out_size; (void)ws_size;

  const float* inputs = (const float*)d_in[0];
  const float* hidden = (const float*)d_in[1];
  const float* Ws1    = (const float*)d_in[2];
  const float* bs1    = (const float*)d_in[3];
  const float* Ws2    = (const float*)d_in[4];
  const float* bs2    = (const float*)d_in[5];
  const float* W1     = (const float*)d_in[6];
  const float* b1     = (const float*)d_in[7];
  const float* Wih    = (const float*)d_in[8];
  const float* Whh    = (const float*)d_in[9];
  const float* bih    = (const float*)d_in[10];
  const float* bhh    = (const float*)d_in[11];
  const float* Wenc   = (const float*)d_in[12];
  const float* benc   = (const float*)d_in[13];
  const float* Wq     = (const float*)d_in[14];
  const float* bq     = (const float*)d_in[15];

  float* q_out = (float*)d_out;                              // [B, A]
  float* h_out = q_out + (size_t)NB * NA;                    // [B, K*H]
  float* a_out = h_out + (size_t)NB * NK * NH;               // [B, K]

  char* ws = (char*)d_ws;
  void* sh_bf = (void*)ws;                                   // B*HS bf16  (4 MB)
  size_t off1 = (size_t)NB * NHS * 2;
  off1 = (off1 + 255) & ~(size_t)255;
  void* x_bf = (void*)(ws + off1);                           // B*K*H bf16 (128 MB)
  size_t off2 = off1 + (size_t)NB * NK * NH * 2;
  off2 = (off2 + 255) & ~(size_t)255;
  float* combined = (float*)(ws + off2);                     // B*E f32   (32 MB)

  // 1) sh = relu(inputs @ Ws1^T + bs1)  -> bf16 scratch
  gemm_wmma_kernel<true, true><<<dim3(NHS / 64, NB / 128, 1), 256, 0, stream>>>(
      inputs, ND, Ws1, bs1, sh_bf, NHS, NB, NHS, ND, 0, 0, 0);

  // 2) alpha = softmax(sh @ Ws2^T + bs2)
  selector_kernel<<<dim3(NB / 8), 256, 0, stream>>>(sh_bf, Ws2, bs2, a_out);

  // 3) x_k = relu(inputs @ W1[k]^T + b1[k])  -> bf16 scratch [B, K*H]
  gemm_wmma_kernel<true, true><<<dim3(NH / 64, NB / 128, NK), 256, 0, stream>>>(
      inputs, ND, W1, b1, x_bf, NK * NH, NB, NH, ND,
      (long)NH * ND, (long)NH, (long)NH);

  // 4) fused GRU -> h_out
  gru_fused_kernel<<<dim3(NH / 64, NB / 64, NK), 256, 0, stream>>>(
      x_bf, hidden, Wih, Whh, bih, bhh, h_out);

  // 5) combined[b,e] = sum_k alpha*(h_k @ Wenc[k]^T + benc[k])
  combine_kernel<<<dim3(NE / 64, NB / 128), 256, 0, stream>>>(
      h_out, a_out, Wenc, benc, combined);

  // 6) q = combined @ Wq^T + bq   (fp32 out, N=32 partial tile)
  gemm_wmma_kernel<false, false><<<dim3(1, NB / 128, 1), 256, 0, stream>>>(
      combined, NE, Wq, bq, q_out, NA, NB, NA, NE, 0, 0, 0);
}
